// MultiHeadLatentAttention_91182155694612
// MI455X (gfx1250) — matrix operand order repaired and checked by reading, not compiled, not run on hardware
//
#include <hip/hip_runtime.h>

// ---------------------------------------------------------------------------
// MI455X (gfx1250) MLA forward: all GEMMs + attention on v_wmma_f32_16x16x32_bf16
// wave32, WGP-mode friendly block sizes, f32 accumulation throughout.
// Flash-attention Q tile is staged into LDS via the Tensor Data Mover (TDM).
// ---------------------------------------------------------------------------

typedef __attribute__((ext_vector_type(16))) __bf16 v16bf;
typedef __attribute__((ext_vector_type(8)))  float  v8f;
typedef __attribute__((ext_vector_type(4)))  __bf16 v4bf;
typedef __attribute__((ext_vector_type(4)))  unsigned int u32x4;
typedef __attribute__((ext_vector_type(8)))  int i32x8;
typedef __attribute__((ext_vector_type(4)))  int i32x4;

union FragBF { v16bf f; uint4 u[2]; };

#define WMMA_BF16(a, b, c) \
  __builtin_amdgcn_wmma_f32_16x16x32_bf16(false, (a), false, (b), (short)0, (c), false, false)

// Problem constants (from reference)
#define BB 2
#define SS 2048
#define EMB 2048
#define NH 16
#define HD 128
#define KV_C 512
#define Q_C 1536
#define MROWS (BB * SS)   // 4096

// --- A fragment: 16x32 bf16 tile, memory row-major [16, ld], lane r=lane&15,
//     k chunks at (lane>>4)*8 and +16 (ISA 7.12.2 16-bit A layout). Two b128 loads.
__device__ __forceinline__ v16bf load_frag_a(const __bf16* base, int ld) {
  const int lane = threadIdx.x & 31;
  const int r    = lane & 15;
  const int koff = (lane >> 4) * 8;
  const __bf16* p = base + (size_t)r * ld + koff;
  FragBF fr;
  fr.u[0] = *(const uint4*)(p);
  fr.u[1] = *(const uint4*)(p + 16);
  return fr.f;
}

// --- B fragment from B^T stored row-major [N, K] (i.e. weight [out,in] layout):
//     lane n=lane&15, k chunks at (lane>>4)*8 and +16 (same k order as A frag).
__device__ __forceinline__ v16bf load_frag_b(const __bf16* base, int ld) {
  const int lane = threadIdx.x & 31;
  const int n  = lane & 15;
  const int kb = (lane >> 4) * 8;
  const __bf16* p = base + (size_t)n * ld + kb;
  FragBF fr;
  fr.u[0] = *(const uint4*)(p);
  fr.u[1] = *(const uint4*)(p + 16);
  return fr.f;
}

// ---------------------------------------------------------------------------
// TDM: 2D tile load Global -> LDS (CDNA5 ISA).
// Descriptor groups built per the D# bit layout; issued by one wave,
// completion via s_wait_tensorcnt, then workgroup barrier.
// ---------------------------------------------------------------------------
#if defined(__has_builtin)
#if __has_builtin(__builtin_amdgcn_tensor_load_to_lds) && \
    __has_builtin(__builtin_amdgcn_s_wait_tensorcnt)
#define HAVE_TDM 1
#endif
#endif
#ifndef HAVE_TDM
#define HAVE_TDM 0
#endif

#if HAVE_TDM
__device__ __forceinline__ void tdm_load_2d_to_lds(
    unsigned lds_byte_off, const void* gptr,
    unsigned tile_w_elems, unsigned tile_h_rows,
    unsigned tensor_w_elems, unsigned tensor_h_rows,
    unsigned row_stride_elems) {
  const unsigned long long ga = (unsigned long long)gptr;  // byte address
  u32x4 g0;
  g0[0] = 1u;                                   // count=1, user mode, no gather
  g0[1] = lds_byte_off;                         // lds_addr
  g0[2] = (unsigned)(ga & 0xFFFFFFFFu);         // global_addr[31:0]
  g0[3] = (unsigned)((ga >> 32) & 0x1FFFFFFu)   // global_addr[56:32]
          | (2u << 30);                         // type = 2 ("image")
  i32x8 g1;
  g1[0] = (int)(1u << 16);                      // data_size=1 (2 bytes), wg_mask=0
  g1[1] = (int)((tensor_w_elems & 0xFFFFu) << 16);             // dim0[15:0]
  g1[2] = (int)(((tensor_w_elems >> 16) & 0xFFFFu) |
                ((tensor_h_rows & 0xFFFFu) << 16));            // dim0[31:16] | dim1[15:0]
  g1[3] = (int)(((tensor_h_rows >> 16) & 0xFFFFu) |
                ((tile_w_elems & 0xFFFFu) << 16));             // dim1[31:16] | tile_dim0
  g1[4] = (int)(tile_h_rows & 0xFFFFu);                        // tile_dim1 (tile_dim2=0)
  g1[5] = (int)row_stride_elems;                               // dim0_stride[31:0]
  g1[6] = 0;                                                   // stride hi, dim1_stride lo
  g1[7] = 0;
  const i32x4 gz = {0, 0, 0, 0};
#if __clang_major__ >= 23
  const i32x8 gz8 = {0, 0, 0, 0, 0, 0, 0, 0};
  __builtin_amdgcn_tensor_load_to_lds(g0, g1, gz, gz, gz8, 0);
#else
  __builtin_amdgcn_tensor_load_to_lds(g0, g1, gz, gz, 0);
#endif
}
#endif  // HAVE_TDM

// ---------------------------------------------------------------------------
// f32 -> bf16 conversion, 4 elements per thread (all sizes are multiples of 4)
// ---------------------------------------------------------------------------
__global__ __launch_bounds__(256) void cvt_bf16(const float* __restrict__ in,
                                                __bf16* __restrict__ out, int n4) {
  int i = blockIdx.x * 256 + threadIdx.x;
  if (i < n4) {
    float4 v = ((const float4*)in)[i];
    v4bf o;
    o[0] = (__bf16)v.x; o[1] = (__bf16)v.y; o[2] = (__bf16)v.z; o[3] = (__bf16)v.w;
    ((v4bf*)out)[i] = o;
  }
}

// ---------------------------------------------------------------------------
// GEMM: C[M,N] = A[M,K] (row-major bf16) * W[N,K]^T (weight layout, bf16)
// Block: 256 threads = 8 wave32s (4 x 2), block tile 256x128, wave tile 64x64.
// M % 256 == 0, N % 128 == 0, K % 32 == 0 for all five calls.
// ---------------------------------------------------------------------------
template <bool OUT_BF16>
__global__ __launch_bounds__(256) void gemm_wmma(const __bf16* __restrict__ A,
                                                 const __bf16* __restrict__ W,
                                                 void* __restrict__ Cout,
                                                 int M, int N, int K) {
  const int wid  = threadIdx.x >> 5;
  const int lane = threadIdx.x & 31;
  const int wm = wid >> 1;          // 0..3
  const int wn = wid & 1;           // 0..1
  const int row0 = blockIdx.y * 256 + wm * 64;
  const int col0 = blockIdx.x * 128 + wn * 64;

  v8f acc[4][4] = {};
  for (int k = 0; k < K; k += 32) {
    v16bf a[4], b[4];
#pragma unroll
    for (int mt = 0; mt < 4; ++mt)
      a[mt] = load_frag_a(A + (size_t)(row0 + mt * 16) * K + k, K);
#pragma unroll
    for (int nt = 0; nt < 4; ++nt)
      b[nt] = load_frag_b(W + (size_t)(col0 + nt * 16) * K + k, K);
    // next-k prefetch hint (global_prefetch_b8) to keep HBM/L2 ahead of XDL
    if (k + 32 < K) {
      __builtin_prefetch(A + (size_t)row0 * K + k + 32, 0, 3);
      __builtin_prefetch(W + (size_t)col0 * K + k + 32, 0, 3);
    }
#pragma unroll
    for (int mt = 0; mt < 4; ++mt)
#pragma unroll
      for (int nt = 0; nt < 4; ++nt)
        acc[mt][nt] = WMMA_BF16(a[mt], b[nt], acc[mt][nt]);
  }

  const int hv = lane >> 4;
  const int nn = lane & 15;
#pragma unroll
  for (int mt = 0; mt < 4; ++mt)
#pragma unroll
    for (int nt = 0; nt < 4; ++nt)
#pragma unroll
      for (int i = 0; i < 8; ++i) {
        const int r = row0 + mt * 16 + i + 8 * hv;   // C-layout: m = vgpr + 8*half
        const int c = col0 + nt * 16 + nn;           // n = lane & 15
        const float v = acc[mt][nt][i];
        if (OUT_BF16) ((__bf16*)Cout)[(size_t)r * N + c] = (__bf16)v;
        else          ((float*)Cout)[(size_t)r * N + c]  = v;
      }
}

// ---------------------------------------------------------------------------
// Causal flash attention.
//   qbuf  : [B*S, NH*HD]  bf16 (col = h*128 + d)
//   kvbuf : [B*S, 2*NH*HD] bf16 (K at h*128+d, V at 2048 + h*128+d)
//   obuf  : [B*S, NH*HD]  bf16
// Block: 128 threads = 4 waves; Q tile 64 rows (16 per wave), KV tile 64.
// Q tile staged once per block into LDS via the Tensor Data Mover.
// Per kv-tile per wave: 16 WMMA (QK^T) + 16 WMMA (PV).
// ---------------------------------------------------------------------------
__global__ __launch_bounds__(128) void mla_flash(const __bf16* __restrict__ qbuf,
                                                 const __bf16* __restrict__ kvbuf,
                                                 __bf16* __restrict__ obuf,
                                                 float scale) {
  constexpr int D = HD;       // 128
  constexpr int T = 64;       // kv tile
  constexpr int QT = 64;      // q tile
  const int qt = blockIdx.x;  // q tile index (S/64 = 32)
  const int h  = blockIdx.y;
  const int b  = blockIdx.z;

  const int wid  = threadIdx.x >> 5;
  const int lane = threadIdx.x & 31;
  const int hv = lane >> 4;
  const int ln = lane & 15;

  __shared__ __align__(16) __bf16 lds_q[QT * D];   // Q tile:   [q][d], 16 KB
  __shared__ __align__(16) __bf16 lds_vt[D * T];   // V^T tile: [d][kv], 16 KB
  __shared__ __align__(16) __bf16 lds_p[QT * T];   // P tile:   [q][kv],  8 KB

  const size_t rowQ  = (size_t)NH * HD;      // 2048
  const size_t rowKV = (size_t)2 * NH * HD;  // 4096
  const int q0 = qt * QT;
  const size_t rbase = (size_t)b * SS;       // row base in flattened [B*S]

  // ---- stage the Q tile into LDS (TDM DMA if available) ----
  const __bf16* qtile = qbuf + (rbase + q0) * rowQ + h * HD;
#if HAVE_TDM
  if (wid == 0) {
    tdm_load_2d_to_lds((unsigned)(size_t)&lds_q[0], qtile,
                       /*tile_w*/ D, /*tile_h*/ QT,
                       /*tensor_w*/ (unsigned)rowQ, /*tensor_h*/ MROWS,
                       /*row_stride*/ (unsigned)rowQ);
    __builtin_amdgcn_s_wait_tensorcnt(0);
  }
#else
  for (int idx = threadIdx.x; idx < QT * D; idx += 128) {
    const int r = idx >> 7;       // / 128
    const int d = idx & 127;
    lds_q[r * D + d] = qtile[(size_t)r * rowQ + d];
  }
#endif

  float mrow[8], lrow[8];
#pragma unroll
  for (int i = 0; i < 8; ++i) { mrow[i] = -1e30f; lrow[i] = 0.f; }
  v8f o[8] = {};

  for (int t = 0; t <= qt; ++t) {
    const int kv0 = t * T;
    __syncthreads();  // Q tile visible (t==0); protect lds_vt from prior PV reads
    // Stage V^T into LDS: lds_vt[d][kv] = V[kv][d]
    for (int idx = threadIdx.x; idx < T * D; idx += 128) {
      const int kv = idx >> 7;      // / 128
      const int d  = idx & 127;
      lds_vt[d * T + kv] =
          kvbuf[(rbase + kv0 + kv) * rowKV + (size_t)(NH * HD) + h * HD + d];
    }
    __syncthreads();

    // ---- S = Q K^T (scaled later). K buffer is [kv, d] == B^T layout. ----
    v8f s[4] = {};
#pragma unroll
    for (int k = 0; k < D; k += 32) {
      const v16bf aq = load_frag_a(lds_q + wid * 16 * D + k, D);
#pragma unroll
      for (int nt = 0; nt < 4; ++nt) {
        const v16bf bk = load_frag_b(
            kvbuf + (rbase + kv0 + nt * 16) * rowKV + h * HD + k, (int)rowKV);
        s[nt] = WMMA_BF16(aq, bk, s[nt]);
      }
    }

    // ---- scale + causal mask + online softmax (C-layout aware) ----
    const bool diag = (t == qt);
    float rmax[8];
#pragma unroll
    for (int i = 0; i < 8; ++i) {
      const int qrow = q0 + wid * 16 + i + 8 * hv;
      float mx = -1e30f;
#pragma unroll
      for (int nt = 0; nt < 4; ++nt) {
        float v = s[nt][i] * scale;
        if (diag && (kv0 + nt * 16 + ln) > qrow) v = -1e30f;
        s[nt][i] = v;
        mx = fmaxf(mx, v);
      }
      // row lives across the 16 lanes of this half-wave
#pragma unroll
      for (int m = 8; m >= 1; m >>= 1) mx = fmaxf(mx, __shfl_xor(mx, m, 32));
      rmax[i] = mx;
    }
#pragma unroll
    for (int i = 0; i < 8; ++i) {
      const float mnew  = fmaxf(mrow[i], rmax[i]);
      const float alpha = __expf(mrow[i] - mnew);
      const int prow = wid * 16 + i + 8 * hv;   // local q row within 64
      float rsum = 0.f;
#pragma unroll
      for (int nt = 0; nt < 4; ++nt) {
        const float p = __expf(s[nt][i] - mnew);
        rsum += p;
        lds_p[prow * T + nt * 16 + ln] = (__bf16)p;  // own rows only
      }
#pragma unroll
      for (int m = 8; m >= 1; m >>= 1) rsum += __shfl_xor(rsum, m, 32);
      lrow[i] = lrow[i] * alpha + rsum;
      mrow[i] = mnew;
#pragma unroll
      for (int dt = 0; dt < 8; ++dt) o[dt][i] *= alpha;
    }

    // ---- O += P V  (P from LDS as A-frags, V^T from LDS as B-frags) ----
    // per-wave LDS ordering (DScnt) guarantees our P stores precede these loads
#pragma unroll
    for (int ks = 0; ks < T; ks += 32) {
      const v16bf ap = load_frag_a(lds_p + wid * 16 * T + ks, T);
#pragma unroll
      for (int dt = 0; dt < 8; ++dt) {
        const v16bf bv = load_frag_b(lds_vt + dt * 16 * T + ks, T);
        o[dt] = WMMA_BF16(ap, bv, o[dt]);
      }
    }
  }

  // ---- normalize and write out ----
#pragma unroll
  for (int dt = 0; dt < 8; ++dt)
#pragma unroll
    for (int i = 0; i < 8; ++i) {
      const size_t r = rbase + q0 + wid * 16 + i + 8 * hv;
      obuf[r * rowQ + h * HD + dt * 16 + ln] = (__bf16)(o[dt][i] / lrow[i]);
    }
}

// ---------------------------------------------------------------------------
// Host launcher
// ---------------------------------------------------------------------------
extern "C" void kernel_launch(void* const* d_in, const int* in_sizes, int n_in,
                              void* d_out, int out_size, void* d_ws, size_t ws_size,
                              hipStream_t stream) {
  const float* x      = (const float*)d_in[0];   // [B,S,EMB]
  const float* w_dq   = (const float*)d_in[1];   // [Q_C, EMB]
  const float* w_uq   = (const float*)d_in[2];   // [EMB, Q_C]
  const float* w_dkv  = (const float*)d_in[3];   // [KV_C, EMB]
  const float* w_ukv  = (const float*)d_in[4];   // [2*EMB, KV_C]
  const float* w_out  = (const float*)d_in[5];   // [EMB, EMB]
  float* out = (float*)d_out;                    // [B,S,EMB] f32

  __bf16* ws = (__bf16*)d_ws;
  size_t off = 0;
  auto alloc = [&](size_t n) { __bf16* p = ws + off; off += n; return p; };

  __bf16* xb    = alloc((size_t)MROWS * EMB);
  __bf16* wdqb  = alloc((size_t)Q_C * EMB);
  __bf16* wuqb  = alloc((size_t)EMB * Q_C);
  __bf16* wdkvb = alloc((size_t)KV_C * EMB);
  __bf16* wukvb = alloc((size_t)2 * EMB * KV_C);
  __bf16* woutb = alloc((size_t)EMB * EMB);
  __bf16* qlat  = alloc((size_t)MROWS * Q_C);
  __bf16* kvlat = alloc((size_t)MROWS * KV_C);
  __bf16* qb    = alloc((size_t)MROWS * EMB);
  __bf16* kvb   = alloc((size_t)MROWS * 2 * EMB);
  __bf16* ob    = alloc((size_t)MROWS * EMB);
  (void)ws_size; (void)n_in; (void)in_sizes; (void)out_size;

  auto cvt = [&](const float* src, __bf16* dst, size_t n) {
    const int n4 = (int)(n / 4);
    cvt_bf16<<<(n4 + 255) / 256, 256, 0, stream>>>(src, dst, n4);
  };
  cvt(x,     xb,    (size_t)MROWS * EMB);
  cvt(w_dq,  wdqb,  (size_t)Q_C * EMB);
  cvt(w_uq,  wuqb,  (size_t)EMB * Q_C);
  cvt(w_dkv, wdkvb, (size_t)KV_C * EMB);
  cvt(w_ukv, wukvb, (size_t)2 * EMB * KV_C);
  cvt(w_out, woutb, (size_t)EMB * EMB);

  // q_lat = x @ w_dq^T            : [4096,1536]
  gemm_wmma<true><<<dim3(Q_C / 128, MROWS / 256), 256, 0, stream>>>(
      xb, wdqb, qlat, MROWS, Q_C, EMB);
  // kv_lat = x @ w_dkv^T          : [4096,512]
  gemm_wmma<true><<<dim3(KV_C / 128, MROWS / 256), 256, 0, stream>>>(
      xb, wdkvb, kvlat, MROWS, KV_C, EMB);
  // q = q_lat @ w_uq^T            : [4096,2048]
  gemm_wmma<true><<<dim3(EMB / 128, MROWS / 256), 256, 0, stream>>>(
      qlat, wuqb, qb, MROWS, EMB, Q_C);
  // kv = kv_lat @ w_ukv^T         : [4096,4096]
  gemm_wmma<true><<<dim3((2 * EMB) / 128, MROWS / 256), 256, 0, stream>>>(
      kvlat, wukvb, kvb, MROWS, 2 * EMB, KV_C);

  // causal flash attention        : ob [4096,2048]
  const float scale = 1.0f / sqrtf((float)HD);
  mla_flash<<<dim3(SS / 64, NH, BB), 128, 0, stream>>>(qb, kvb, ob, scale);

  // out = attn_out @ w_out^T      : f32 [4096,2048]
  gemm_wmma<false><<<dim3(EMB / 128, MROWS / 256), 256, 0, stream>>>(
      ob, woutb, out, MROWS, EMB, EMB);
}
